// LongContextAttention_42984032698488
// MI455X (gfx1250) — matrix operand order repaired and checked by reading, not compiled, not run on hardware
//
#include <hip/hip_runtime.h>

typedef __attribute__((ext_vector_type(16))) __bf16 bf16x16;
typedef __attribute__((ext_vector_type(8)))  __bf16 bf16x8;
typedef __attribute__((ext_vector_type(2)))  __bf16 bf16x2;
typedef __attribute__((ext_vector_type(8)))  float  f32x8;
typedef __attribute__((ext_vector_type(8)))  unsigned int u32x8;

#define B_ 2
#define S_ 2048
#define H_ 16
#define D_ 128
#define BM 128      // query rows per workgroup
#define WM 16       // query rows per wave
#define BN 32       // key rows per iteration
#define NWAVES 8

__launch_bounds__(256, 1)
__global__ void fa_fwd_causal(const float* __restrict__ Q,
                              const float* __restrict__ K,
                              const float* __restrict__ V,
                              float* __restrict__ O) {
  // Double-buffered staging:
  // K block bf16, row-major [key][d]   -> A-fragments for S^T = K * Q^T
  __shared__ __align__(64) __bf16 sK[2][BN][D_];    // 16 KB
  // V block bf16, transposed [d][key]  -> A-fragments for O^T = V^T * P
  __shared__ __align__(64) __bf16 sVt[2][D_][BN];   // 16 KB

  const int tid  = threadIdx.x;
  const int wave = tid >> 5;
  const int lane = tid & 31;
  const int ln16 = lane & 15;
  const int hi   = lane >> 4;          // lane half

  const int qtiles = S_ / BM;          // 16
  const int bh     = blockIdx.x / qtiles;
  const int qtile  = blockIdx.x % qtiles;
  const int b = bh / H_;
  const int h = bh % H_;

  const int q0w = qtile * BM + wave * WM;   // wave's first query row
  const int qi  = q0w + ln16;               // this lane's query (N column)
  // log2(e)/sqrt(128) folded into Q so softmax uses raw v_exp_f32 (exp2)
  const float qscale = 0.08838834764831845f * 1.44269504088896340f;

  // ---- Q as B-fragments (resident): lane = query col N=ln16,
  //      element i -> d = 32*kb + 16*hi + i  (contiguous 16 floats) ----------
  bf16x16 qb[4];
  {
    const float* qrow = Q + (((size_t)b * S_ + qi) * H_ + h) * D_;
    #pragma unroll
    for (int kb = 0; kb < 4; ++kb) {
      const float4* qp = (const float4*)(qrow + kb * 32 + hi * 16);
      #pragma unroll
      for (int c = 0; c < 4; ++c) {
        float4 x = qp[c];
        qb[kb][4*c + 0] = (__bf16)(x.x * qscale);
        qb[kb][4*c + 1] = (__bf16)(x.y * qscale);
        qb[kb][4*c + 2] = (__bf16)(x.z * qscale);
        qb[kb][4*c + 3] = (__bf16)(x.w * qscale);
      }
    }
  }

  // O^T accumulators: tile j covers d = 16j..16j+15; element r -> d = 16j+8hi+r
  f32x8 oacc[8];
  const f32x8 fzero = {0.f,0.f,0.f,0.f,0.f,0.f,0.f,0.f};
  #pragma unroll
  for (int j = 0; j < 8; ++j) oacc[j] = fzero;

  float mrun = -1e30f, lrun = 0.0f;    // per-query scalars (replicated in pair)

  // staging prefetch registers (fp32, converted late to overlap loads/WMMA)
  float krf[16], vrf0[8], vrf1[8];
  const int srow  = tid >> 3;          // 0..31       (K staging)
  const int scol0 = (tid & 7) * 16;    // 0..112
  const int skr   = (tid >> 4) * 2;    // 0,2,..,30   (V staging)
  const int sd0   = (tid & 15) * 8;    // 0..120

  auto stage_load = [&](int k0) {
    const float4* kp = (const float4*)(K + (((size_t)b * S_ + (k0 + srow)) * H_ + h) * D_ + scol0);
    #pragma unroll
    for (int c = 0; c < 4; ++c) {
      float4 x = kp[c];
      krf[4*c+0]=x.x; krf[4*c+1]=x.y; krf[4*c+2]=x.z; krf[4*c+3]=x.w;
    }
    const float* v0p = V + (((size_t)b * S_ + (k0 + skr)) * H_ + h) * D_ + sd0;
    const float* v1p = v0p + (size_t)H_ * D_;
    float4 a0 = ((const float4*)v0p)[0], a1 = ((const float4*)v0p)[1];
    float4 b0 = ((const float4*)v1p)[0], b1 = ((const float4*)v1p)[1];
    vrf0[0]=a0.x; vrf0[1]=a0.y; vrf0[2]=a0.z; vrf0[3]=a0.w;
    vrf0[4]=a1.x; vrf0[5]=a1.y; vrf0[6]=a1.z; vrf0[7]=a1.w;
    vrf1[0]=b0.x; vrf1[1]=b0.y; vrf1[2]=b0.z; vrf1[3]=b0.w;
    vrf1[4]=b1.x; vrf1[5]=b1.y; vrf1[6]=b1.z; vrf1[7]=b1.w;
  };
  auto stage_store = [&](int buf) {
    bf16x8 lo8, hi8;
    #pragma unroll
    for (int i = 0; i < 8; ++i) { lo8[i]=(__bf16)krf[i]; hi8[i]=(__bf16)krf[8+i]; }
    *(bf16x8*)&sK[buf][srow][scol0]     = lo8;
    *(bf16x8*)&sK[buf][srow][scol0 + 8] = hi8;
    #pragma unroll
    for (int i = 0; i < 8; ++i) {
      bf16x2 pkv; pkv[0] = (__bf16)vrf0[i]; pkv[1] = (__bf16)vrf1[i];
      *(bf16x2*)&sVt[buf][sd0 + i][skr] = pkv;
    }
  };

  const int nkb = (qtile + 1) * (BM / BN);   // causal key blocks for this WG

  // prologue: stage block 0 into buffer 0
  stage_load(0);
  stage_store(0);

  for (int kbk = 0; kbk < nkb; ++kbk) {
    const int k0  = kbk * BN;
    const int cur = kbk & 1;
    const bool have_next = (kbk + 1) < nkb;

    __syncthreads();                       // buffer `cur` ready for all waves

    if (have_next) stage_load(k0 + BN);    // issue next block's HBM loads now

    if (k0 <= q0w + (WM - 1)) {            // wave-uniform causal visibility
      // -------- S^T tiles: st0 = keys k0..k0+15, st1 = keys k0+16..k0+31 ---
      f32x8 st0 = fzero, st1 = fzero;
      #pragma unroll
      for (int kb = 0; kb < 4; ++kb) {
        // A 16x32: lane row M=ln16 (key), K(d) = 16*hi + 0..15
        bf16x8 a0l = *(const bf16x8*)&sK[cur][ln16     ][kb*32 + hi*16];
        bf16x8 a0h = *(const bf16x8*)&sK[cur][ln16     ][kb*32 + hi*16 + 8];
        bf16x8 a1l = *(const bf16x8*)&sK[cur][16 + ln16][kb*32 + hi*16];
        bf16x8 a1h = *(const bf16x8*)&sK[cur][16 + ln16][kb*32 + hi*16 + 8];
        bf16x16 ka0 = __builtin_shufflevector(a0l, a0h, 0,1,2,3,4,5,6,7,8,9,10,11,12,13,14,15);
        bf16x16 ka1 = __builtin_shufflevector(a1l, a1h, 0,1,2,3,4,5,6,7,8,9,10,11,12,13,14,15);
        st0 = __builtin_amdgcn_wmma_f32_16x16x32_bf16(false, ka0, false, qb[kb],
                                                      (short)0, st0, false, false);
        st1 = __builtin_amdgcn_wmma_f32_16x16x32_bf16(false, ka1, false, qb[kb],
                                                      (short)0, st1, false, false);
      }

      // -------- in-lane online softmax (base-2 domain) ---------------------
      float a[16];
      if (k0 + (BN - 1) <= q0w) {
        // interior block: fully visible, no mask arithmetic at all
        #pragma unroll
        for (int r = 0; r < 8; ++r) { a[r] = st0[r]; a[r + 8] = st1[r]; }
      } else {
        // the single diagonal block of this wave
        #pragma unroll
        for (int r = 0; r < 8; ++r) {
          const int ki = k0 + hi * 8 + r;     // local keys: r+8hi, 16+r+8hi
          a[r]     = (ki      <= qi) ? st0[r] : -1e30f;
          a[r + 8] = (ki + 16 <= qi) ? st1[r] : -1e30f;
        }
      }
      float rm = a[0];
      #pragma unroll
      for (int i = 1; i < 16; ++i) rm = fmaxf(rm, a[i]);
      rm = fmaxf(rm, __shfl_xor(rm, 16));     // combine lane pair -> full row
      const float mnew = fmaxf(mrun, rm);
      const float corr = __builtin_amdgcn_exp2f(mrun - mnew);
      float p[16];
      float rs = 0.0f;
      #pragma unroll
      for (int i = 0; i < 16; ++i) { p[i] = __builtin_amdgcn_exp2f(a[i] - mnew); rs += p[i]; }
      rs += __shfl_xor(rs, 16);
      lrun = lrun * corr + rs;
      mrun = mnew;
      #pragma unroll
      for (int j = 0; j < 8; ++j) oacc[j] *= corr;

      // -------- P^T (C-layout) -> P B-fragment via one xor-16 exchange -----
      // B layout: element i -> key = i + 16*hi, column N = query ln16
      unsigned int d0w[4], d1w[4];
      #pragma unroll
      for (int k = 0; k < 4; ++k) {
        bf16x2 t0; t0[0] = (__bf16)p[2*k];     t0[1] = (__bf16)p[2*k + 1];
        bf16x2 t1; t1[0] = (__bf16)p[8 + 2*k]; t1[1] = (__bf16)p[8 + 2*k + 1];
        d0w[k] = __builtin_bit_cast(unsigned int, t0);
        d1w[k] = __builtin_bit_cast(unsigned int, t1);
      }
      u32x8 pw;
      #pragma unroll
      for (int k = 0; k < 4; ++k) {
        unsigned int o0 = (unsigned int)__shfl_xor((int)d0w[k], 16);
        unsigned int o1 = (unsigned int)__shfl_xor((int)d1w[k], 16);
        pw[k]     = hi ? o1      : d0w[k];   // keys 16*hi + 0..7
        pw[k + 4] = hi ? d1w[k]  : o0;       // keys 16*hi + 8..15
      }
      bf16x16 pb = __builtin_bit_cast(bf16x16, pw);

      // -------- O^T += V^T x P ---------------------------------------------
      #pragma unroll
      for (int j = 0; j < 8; ++j) {
        // A 16x32: lane row M=ln16 -> d = 16j+ln16, K(keys) = 16hi + 0..15
        bf16x8 vl = *(const bf16x8*)&sVt[cur][j*16 + ln16][hi*16];
        bf16x8 vh = *(const bf16x8*)&sVt[cur][j*16 + ln16][hi*16 + 8];
        bf16x16 va = __builtin_shufflevector(vl, vh, 0,1,2,3,4,5,6,7,8,9,10,11,12,13,14,15);
        oacc[j] = __builtin_amdgcn_wmma_f32_16x16x32_bf16(false, va, false, pb,
                                                          (short)0, oacc[j],
                                                          false, false);
      }
    }

    if (have_next) stage_store(cur ^ 1);   // convert + fill the other buffer
  }

  // ---------------- epilogue: O[q, d] = acc / l, contiguous b128 stores ----
  const float inv = 1.0f / lrun;
  float* orow = O + (((size_t)b * S_ + qi) * H_ + h) * D_ + hi * 8;
  #pragma unroll
  for (int j = 0; j < 8; ++j) {
    float4 lo = {oacc[j][0]*inv, oacc[j][1]*inv, oacc[j][2]*inv, oacc[j][3]*inv};
    float4 hh = {oacc[j][4]*inv, oacc[j][5]*inv, oacc[j][6]*inv, oacc[j][7]*inv};
    ((float4*)(orow + j*16))[0]     = lo;
    ((float4*)(orow + j*16 + 4))[0] = hh;
  }
}

extern "C" void kernel_launch(void* const* d_in, const int* in_sizes, int n_in,
                              void* d_out, int out_size, void* d_ws, size_t ws_size,
                              hipStream_t stream) {
  (void)in_sizes; (void)n_in; (void)out_size; (void)d_ws; (void)ws_size;
  const float* Q = (const float*)d_in[0];
  const float* K = (const float*)d_in[1];
  const float* V = (const float*)d_in[2];
  float* O = (float*)d_out;
  const int qtiles = S_ / BM;                 // 16
  dim3 grid(B_ * H_ * qtiles);                // 512 workgroups
  dim3 block(256);                            // 8 waves (wave32)
  fa_fwd_causal<<<grid, block, 0, stream>>>(Q, K, V, O);
}
